// Kipfblock_7748121002165
// MI455X (gfx1250) — matrix operand order repaired and checked by reading, not compiled, not run on hardware
//
#include <hip/hip_runtime.h>

#define NNODES 100000
#define NEDGES 1000000
#define NF 64
#define KCHEB 8

typedef __attribute__((ext_vector_type(16))) _Float16 v16h;
typedef __attribute__((ext_vector_type(8)))  float    v8f;

__device__ __forceinline__ void atomicAddF32(float* p, float v) {
    __hip_atomic_fetch_add(p, v, __ATOMIC_RELAXED, __HIP_MEMORY_SCOPE_AGENT);
}

// ---------------------------------------------------------------- utilities
__global__ void zero_f32_kernel(float* __restrict__ p, int n) {
    int i = blockIdx.x * blockDim.x + threadIdx.x;
    if (i < n) p[i] = 0.0f;
}

// neg_copy may run in place (dst == src), so no restrict here.
__global__ void neg_copy_kernel(float4* dst, const float4* src, int n4) {
    int i = blockIdx.x * blockDim.x + threadIdx.x;
    if (i >= n4) return;
    float4 v = src[i];
    v.x = -v.x; v.y = -v.y; v.z = -v.z; v.w = -v.w;
    dst[i] = v;
}

// ---------------------------------------------------------------- graph prep
__global__ void edge_prep_kernel(const long long* __restrict__ ei,
                                 int* __restrict__ src, int* __restrict__ dst,
                                 float* __restrict__ deg, int nE) {
    int e = blockIdx.x * blockDim.x + threadIdx.x;
    if (e >= nE) return;
    int s = (int)ei[e];
    int d = (int)ei[(size_t)nE + e];
    src[e] = s;
    dst[e] = d;
    atomicAddF32(&deg[s], 1.0f);
}

__global__ void finalize_dis_kernel(float* __restrict__ deg, int n) {
    int i = blockIdx.x * blockDim.x + threadIdx.x;
    if (i >= n) return;
    float d = deg[i];
    deg[i] = (d > 0.0f) ? rsqrtf(fmaxf(d, 1.0f)) : 0.0f;
}

__global__ void norm_kernel(const int* __restrict__ src, const int* __restrict__ dst,
                            const float* __restrict__ dis, float* __restrict__ norm, int nE) {
    int e = blockIdx.x * blockDim.x + threadIdx.x;
    if (e >= nE) return;
    norm[e] = -dis[src[e]] * dis[dst[e]];
}

// ------------------------------------------------ edge gather/scatter (prop)
// One wave32 per edge; each lane moves 2 features. T += scale*norm[e]*H[src].
__global__ void __launch_bounds__(256)
scatter_kernel(const int* __restrict__ src, const int* __restrict__ dst,
               const float* __restrict__ norm, const float* __restrict__ H,
               float* __restrict__ T, float scale, int nE) {
    int idx  = blockIdx.x * blockDim.x + threadIdx.x;
    int e    = idx >> 5;                 // wave-uniform
    if (e >= nE) return;
    int lane = threadIdx.x & 31;
    float coef = scale * norm[e];
    int s = src[e];
    int d = dst[e];
    float2 h = *reinterpret_cast<const float2*>(H + (size_t)s * NF + lane * 2);
    float* tp = T + (size_t)d * NF + lane * 2;
    atomicAddF32(tp + 0, coef * h.x);
    atomicAddF32(tp + 1, coef * h.y);
}

// ------------------------------------------------------------- WMMA GEMM
// out[16x64 tile] (+)= T[16x64] @ W[64x64]   using v_wmma_f32_16x16x32_f16.
// mode 0: out = acc + bias   (k=0)
// mode 1: out += acc         (k=1..6)
// mode 2: out = relu(out+acc) (k=7)
__global__ void __launch_bounds__(256)
cheb_gemm_kernel(const float* __restrict__ T, const float* __restrict__ W,
                 const float* __restrict__ bias, float* __restrict__ out,
                 int nRows, int mode) {
    const int wave = blockIdx.x * (blockDim.x >> 5) + (threadIdx.x >> 5);
    const int lane = threadIdx.x & 31;
    const int m0   = wave << 4;
    if (m0 >= nRows) return;            // wave-uniform: EXEC all-ones for WMMA
    const int mrow = lane & 15;
    const int hi   = lane >> 4;         // 0: lanes 0-15, 1: lanes 16-31

    v8f c[4] = {};
    const float* rowp = T + (size_t)(m0 + mrow) * NF;

#pragma unroll
    for (int ksel = 0; ksel < 2; ++ksel) {
        // A-matrix 16x32 f16 layout (ISA 7.12.2, dense 16-bit A):
        // lanes 0-15: V0-3 = K 0..7, V4-7 = K 16..23 ; lanes 16-31: +8
        const int kbA = ksel * 32 + hi * 8;
        v16h a;
#pragma unroll
        for (int j = 0; j < 8; ++j) {
            a[j]     = (_Float16)rowp[kbA + j];
            a[8 + j] = (_Float16)rowp[kbA + 16 + j];
        }
        // B-matrix 32x16 f16: same K order as A (lanes 0-15: K 0..7,16..23 ; lanes 16-31: +8)
        const int kbB = ksel * 32 + hi * 8;
#pragma unroll
        for (int nt = 0; nt < 4; ++nt) {
            const int n = nt * 16 + mrow;
            v16h b;
#pragma unroll
            for (int j = 0; j < 8; ++j) {
                b[j]     = (_Float16)W[(kbB + j) * NF + n];
                b[8 + j] = (_Float16)W[(kbB + 16 + j) * NF + n];
            }
            c[nt] = __builtin_amdgcn_wmma_f32_16x16x32_f16(
                false, a, false, b, (short)0, c[nt], false, false);
        }
    }

    // C/D layout: VGPR v -> M = v + hi*8, N = lane%16
#pragma unroll
    for (int nt = 0; nt < 4; ++nt) {
        const int n = nt * 16 + mrow;
#pragma unroll
        for (int v = 0; v < 8; ++v) {
            const int m    = m0 + hi * 8 + v;
            const size_t o = (size_t)m * NF + n;
            const float val = c[nt][v];
            if (mode == 0) {
                out[o] = val + bias[n];
            } else if (mode == 1) {
                out[o] += val;
            } else {
                float r = out[o] + val;
                out[o] = fmaxf(r, 0.0f);
            }
        }
    }
}

// ---------------------------------------------------------------- launcher
extern "C" void kernel_launch(void* const* d_in, const int* in_sizes, int n_in,
                              void* d_out, int out_size, void* d_ws, size_t ws_size,
                              hipStream_t stream) {
    const float*     x    = (const float*)d_in[0];       // [100000,64]
    const long long* ei   = (const long long*)d_in[1];   // [2,1000000] int64
    const float*     W    = (const float*)d_in[2];       // [8,64,64]
    const float*     bias = (const float*)d_in[3];       // [64]
    float*           out  = (float*)d_out;               // [100000,64]

    // Workspace layout (all offsets 256B aligned, total ~64.5 MB)
    char*  ws   = (char*)d_ws;
    float* dis  = (float*)(ws);                                         // 100000 f32 (deg -> deg^-1/2)
    int*   srcA = (int*)  (ws + (size_t)1  * 1024 * 1024);              // 1M i32
    int*   dstA = (int*)  (ws + (size_t)5  * 1024 * 1024);              // 1M i32
    float* nrm  = (float*)(ws + (size_t)9  * 1024 * 1024);              // 1M f32
    float* TA   = (float*)(ws + (size_t)13 * 1024 * 1024);              // 6.4M f32
    float* TB   = (float*)(ws + (size_t)39 * 1024 * 1024);              // 6.4M f32

    const int TPB = 256;
    const int nodeBlocks = (NNODES + TPB - 1) / TPB;
    const int edgeBlocks = (NEDGES + TPB - 1) / TPB;
    const int featElems  = NNODES * NF;                   // 6.4M
    const int featBlocks = (featElems + TPB - 1) / TPB;
    const int feat4Blocks = (featElems / 4 + TPB - 1) / TPB;
    const int scatBlocks = (NEDGES * 32) / TPB;           // 1 wave per edge
    const int gemmBlocks = (NNODES / 16 + 7) / 8;         // 8 waves/block, 16 rows/wave

    // ---- graph normalization
    zero_f32_kernel<<<nodeBlocks, TPB, 0, stream>>>(dis, NNODES);
    edge_prep_kernel<<<edgeBlocks, TPB, 0, stream>>>(ei, srcA, dstA, dis, NEDGES);
    finalize_dis_kernel<<<nodeBlocks, TPB, 0, stream>>>(dis, NNODES);
    norm_kernel<<<edgeBlocks, TPB, 0, stream>>>(srcA, dstA, dis, nrm, NEDGES);

    // ---- k = 0: out = x @ W0 + bias
    cheb_gemm_kernel<<<gemmBlocks, 256, 0, stream>>>(x, W, bias, out, NNODES, 0);

    // ---- k = 1: T1 = prop(x); out += T1 @ W1
    zero_f32_kernel<<<featBlocks, TPB, 0, stream>>>(TA, featElems);
    scatter_kernel<<<scatBlocks, TPB, 0, stream>>>(srcA, dstA, nrm, x, TA, 1.0f, NEDGES);
    cheb_gemm_kernel<<<gemmBlocks, 256, 0, stream>>>(TA, W + NF * NF, bias, out, NNODES, 1);

    // ---- k = 2..7: Tk = 2*prop(T_{k-1}) - T_{k-2}  (acc initialized to -T_{k-2})
    const float* prev = x;   // T_{k-2}
    float*       cur  = TA;  // T_{k-1}
    for (int k = 2; k < KCHEB; ++k) {
        float* nxt = (k & 1) ? TA : TB;   // k=2 -> TB, k=3 -> TA (in-place negate), ...
        neg_copy_kernel<<<feat4Blocks, TPB, 0, stream>>>(
            (float4*)nxt, (const float4*)prev, featElems / 4);
        scatter_kernel<<<scatBlocks, TPB, 0, stream>>>(srcA, dstA, nrm, cur, nxt, 2.0f, NEDGES);
        cheb_gemm_kernel<<<gemmBlocks, 256, 0, stream>>>(
            nxt, W + (size_t)k * NF * NF, bias, out, NNODES, (k == KCHEB - 1) ? 2 : 1);
        prev = cur;
        cur  = nxt;
    }
}
